// HymbaAttention_13692355740247
// MI455X (gfx1250) — matrix operand order repaired and checked by reading, not compiled, not run on hardware
//
#include <hip/hip_runtime.h>
#include <hip/hip_bf16.h>
#include <stdint.h>

// ---------------------------------------------------------------- constants
#define S_LEN 2048
#define D_DIM 2048
#define NH    32
#define NKV   8
#define DK    64
#define DV    64
#define WIN   1024
#define QB    128          // queries per attention workgroup (8 waves x 16)
#define KLP   72           // LDS K tile row stride (bf16), padded vs 64
#define VLP   40           // LDS V tile row stride (bf16), padded vs 32

typedef __bf16 bf16_t;
typedef __attribute__((ext_vector_type(16))) __bf16 v16bf;
typedef __attribute__((ext_vector_type(8)))  __bf16 v8bf;
typedef __attribute__((ext_vector_type(8)))  float  v8f;

// ----------------------------------------------------------------- helpers
static __device__ __forceinline__ bf16_t f2bf(float f) {
  uint32_t u = __builtin_bit_cast(uint32_t, f);
  uint32_t r = u + 0x7FFFu + ((u >> 16) & 1u);      // RNE f32 -> bf16
  uint16_t h = (uint16_t)(r >> 16);
  return __builtin_bit_cast(bf16_t, h);
}

// 16 contiguous bf16 -> B-fragment (element e -> K=e within 16-lane half)
static __device__ __forceinline__ v16bf ld_frag16(const bf16_t* p) {
  v8bf lo = *(const v8bf*)p;
  v8bf hi = *(const v8bf*)(p + 8);
  return __builtin_shufflevector(lo, hi, 0,1,2,3,4,5,6,7,8,9,10,11,12,13,14,15);
}

// A-matrix 16x32 fragment: e -> K=2*kbase+e (contiguous, matches B fragments)
static __device__ __forceinline__ v16bf ld_fragA(const bf16_t* p, int kbase) {
  v8bf lo = *(const v8bf*)(p + 2 * kbase);
  v8bf hi = *(const v8bf*)(p + 2 * kbase + 8);
  return __builtin_shufflevector(lo, hi, 0,1,2,3,4,5,6,7,8,9,10,11,12,13,14,15);
}

static __device__ __forceinline__ v8f wmma_bf16(v16bf a, v16bf b, v8f c) {
  return __builtin_amdgcn_wmma_f32_16x16x32_bf16(false, a, false, b, (short)0, c,
                                                 false, false);
}

// async copy of 16B global -> LDS (gfx1250 GLOBAL_LOAD_ASYNC_TO_LDS_B128,
// tracked with ASYNCcnt; see CDNA5 ISA)
static __device__ __forceinline__ void async_g2l_b128(const bf16_t* g, const bf16_t* l) {
  uint32_t lds = (uint32_t)(uintptr_t)l;            // low 32 bits = LDS byte addr
  uint64_t ga  = (uint64_t)(uintptr_t)g;
  asm volatile("global_load_async_to_lds_b128 %0, %1, off"
               :: "v"(lds), "v"(ga) : "memory");
}

// ------------------------------------------------------------ f32 -> bf16
__global__ void __launch_bounds__(256) cvt_f32_bf16(const float* __restrict__ src,
                                                    bf16_t* __restrict__ dst, int n) {
  int i = blockIdx.x * blockDim.x + threadIdx.x;
  int stride = gridDim.x * blockDim.x;
  for (; i < n; i += stride) dst[i] = f2bf(src[i]);
}

// ---------------------------------------------- GEMM: C[M,N] = A[M,K] * W[N,K]^T
// One wave computes a 32x64 tile (2 A frags x 4 B frags = 8 WMMA / K-step).
__global__ void __launch_bounds__(256) gemm_bf16_nt(const bf16_t* __restrict__ A,
                                                    const bf16_t* __restrict__ W,
                                                    float* __restrict__ C,
                                                    int M, int N, int K) {
  const int lane   = threadIdx.x & 31;
  const int wave   = (blockIdx.x * blockDim.x + threadIdx.x) >> 5;
  const int ntiles = N >> 6;
  const int mtile  = wave / ntiles;
  const int ntile  = wave - mtile * ntiles;
  if (mtile * 32 >= M) return;                      // wave-uniform guard

  const int ln15  = lane & 15;
  const int kbase = (lane >> 4) << 3;
  const int koff  = (lane >> 4) << 4;
  const int col0  = ntile * 64;

  const bf16_t* ap0 = A + (size_t)(mtile * 32 + ln15) * K;
  const bf16_t* ap1 = ap0 + (size_t)16 * K;
  const bf16_t* wp0 = W + (size_t)(col0 + ln15) * K + koff;
  const bf16_t* wp1 = wp0 + (size_t)16 * K;
  const bf16_t* wp2 = wp0 + (size_t)32 * K;
  const bf16_t* wp3 = wp0 + (size_t)48 * K;

  v8f c00 = {}, c01 = {}, c02 = {}, c03 = {};
  v8f c10 = {}, c11 = {}, c12 = {}, c13 = {};
  for (int kk = 0; kk < K; kk += 32) {
    v16bf a0 = ld_fragA(ap0 + kk, kbase);
    v16bf a1 = ld_fragA(ap1 + kk, kbase);
    v16bf b0 = ld_frag16(wp0 + kk);
    v16bf b1 = ld_frag16(wp1 + kk);
    v16bf b2 = ld_frag16(wp2 + kk);
    v16bf b3 = ld_frag16(wp3 + kk);
    c00 = wmma_bf16(a0, b0, c00);  c10 = wmma_bf16(a1, b0, c10);
    c01 = wmma_bf16(a0, b1, c01);  c11 = wmma_bf16(a1, b1, c11);
    c02 = wmma_bf16(a0, b2, c02);  c12 = wmma_bf16(a1, b2, c12);
    c03 = wmma_bf16(a0, b3, c03);  c13 = wmma_bf16(a1, b3, c13);
  }

  const int cm = mtile * 32 + ((lane >> 4) << 3);
  const int cn = col0 + ln15;
#pragma unroll
  for (int r = 0; r < 8; ++r) {
    float* cp = C + (size_t)(cm + r) * N + cn;
    cp[0] = c00[r]; cp[16] = c01[r]; cp[32] = c02[r]; cp[48] = c03[r];
    float* cq = C + (size_t)(cm + 16 + r) * N + cn;
    cq[0] = c10[r]; cq[16] = c11[r]; cq[32] = c12[r]; cq[48] = c13[r];
  }
}

// --------------------------------- per-head RMSNorm + RoPE, write head-major bf16
__global__ void __launch_bounds__(256) rmsnorm_rope(const float* __restrict__ src,
                                                    const float* __restrict__ w,
                                                    const int* __restrict__ pos,
                                                    bf16_t* __restrict__ dst, int H) {
  const int lane = threadIdx.x & 31;
  const int wave = (blockIdx.x * blockDim.x + threadIdx.x) >> 5;
  const int s = wave / H;
  const int h = wave - s * H;
  if (s >= S_LEN) return;

  const float* p = src + ((size_t)s * H + h) * 64;
  float x0 = p[lane], x1 = p[lane + 32];
  float ss = x0 * x0 + x1 * x1;
  ss += __shfl_xor(ss, 1);  ss += __shfl_xor(ss, 2);
  ss += __shfl_xor(ss, 4);  ss += __shfl_xor(ss, 8);
  ss += __shfl_xor(ss, 16);
  float rn = rsqrtf(ss * (1.0f / 64.0f) + 1e-6f);
  float y0 = w[h * 64 + lane]      * x0 * rn;
  float y1 = w[h * 64 + lane + 32] * x1 * rn;

  float invf = __powf(10000.0f, -(float)(2 * lane) * (1.0f / 64.0f));
  float ang  = (float)pos[s] * invf;
  float c  = __cosf(ang);
  float sn = __sinf(ang);

  bf16_t* q = dst + ((size_t)h * S_LEN + s) * 64;
  q[lane]      = f2bf(y0 * c - y1 * sn);
  q[lane + 32] = f2bf(y1 * c + y0 * sn);
}

// ------------------------- V: [S, NKV*64] f32 -> [NKV, 64, S] bf16 (transposed)
__global__ void __launch_bounds__(256) vconvert(const float* __restrict__ Vtmp,
                                                bf16_t* __restrict__ Vt) {
  const int n = NKV * DV * S_LEN;
  int i = blockIdx.x * blockDim.x + threadIdx.x;
  if (i >= n) return;
  int row = i >> 11;                     // kv*64 + d
  int s   = i & (S_LEN - 1);
  Vt[i] = f2bf(Vtmp[(size_t)s * (NKV * DV) + row]);
}

// ----------------- sliding-window flash attention
// Block = 1 head x 128 queries (8 waves x 16). K/V chunks staged to LDS with
// async loads (double buffered); per-wave P round-trips LDS (C- -> A-layout).
// Qa: [NH,S,64]  Ka: [NKV,S,64]  Vt: [NKV,64,S]  Oa: [S, NH*64] bf16
__global__ void __launch_bounds__(256) attn_swa(const bf16_t* __restrict__ Qa,
                                                const bf16_t* __restrict__ Ka,
                                                const bf16_t* __restrict__ Vt,
                                                bf16_t* __restrict__ Oa) {
  __shared__ __align__(16) bf16_t Kl[2][32][KLP];   // 32 keys x 64 dk (padded)
  __shared__ __align__(16) bf16_t Vl[2][64][VLP];   // 64 dv  x 32 k  (padded)
  __shared__ __align__(16) bf16_t Pt[8][16][32];    // per-wave P tile

  const int tid    = threadIdx.x;
  const int lane   = tid & 31;
  const int wavein = tid >> 5;
  const int h   = blockIdx.x >> 4;                  // 16 blocks per head
  const int q0b = (blockIdx.x & 15) << 7;
  const int hk  = h >> 2;                           // GQA group of 4
  const int q0  = q0b + (wavein << 4);

  const int ln15  = lane & 15;
  const int kbase = (lane >> 4) << 3;
  const int koff  = (lane >> 4) << 4;
  const int mrow  = (lane >> 4) << 3;

  // cooperative staging geometry: 256 threads x 16B for K, same for V
  const int krow = tid >> 3, kseg = (tid & 7) << 3;   // K: 32 rows x 128B
  const int vrow = tid >> 2, vseg = (tid & 3) << 3;   // V: 64 rows x 64B
  const bf16_t* kgb = Ka + ((size_t)hk * S_LEN + krow) * 64 + kseg;
  const bf16_t* vgb = Vt + ((size_t)hk * DV + vrow) * S_LEN + vseg;

  // Q fragments (dk 0..31, 32..63)
  const bf16_t* qp = Qa + ((size_t)h * S_LEN + q0 + ln15) * 64;
  v16bf qf0 = ld_fragA(qp, kbase);
  v16bf qf1 = ld_fragA(qp + 32, kbase);

  float rmax[8], rsum[8];
#pragma unroll
  for (int r = 0; r < 8; ++r) { rmax[r] = -1e30f; rsum[r] = 0.0f; }
  v8f o0 = {}, o1 = {}, o2 = {}, o3 = {};

  int kstart = q0b - WIN + 1; if (kstart < 0) kstart = 0;
  kstart &= ~31;
  const int nchunk = (q0b + QB - kstart) >> 5;      // exact (both ends 32-aligned)

  // preload chunk 0
  async_g2l_b128(kgb + (size_t)krow * 0 + kstart, &Kl[0][krow][kseg]);
  async_g2l_b128(vgb + kstart, &Vl[0][vrow][vseg]);

  for (int ci = 0; ci < nchunk; ++ci) {
    const int kc  = kstart + ci * 32;
    const int buf = ci & 1;
    asm volatile("s_wait_asynccnt 0" ::: "memory");
    __syncthreads();                                // LDS chunk visible; prev compute done
    if (ci + 1 < nchunk) {                          // stage next chunk (other buffer)
      async_g2l_b128(kgb + (kc + 32), &Kl[buf ^ 1][krow][kseg]);
      async_g2l_b128(vgb + (kc + 32), &Vl[buf ^ 1][vrow][vseg]);
    }

    // ---- scores: 16 queries x 32 keys, dk reduced over 2 WMMA steps per tile
    const bf16_t* kp0 = &Kl[buf][ln15][koff];
    const bf16_t* kp1 = &Kl[buf][16 + ln15][koff];
    v8f s0 = {}, s1 = {};
    s0 = wmma_bf16(qf0, ld_frag16(kp0),      s0);
    s0 = wmma_bf16(qf1, ld_frag16(kp0 + 32), s0);
    s1 = wmma_bf16(qf0, ld_frag16(kp1),      s1);
    s1 = wmma_bf16(qf1, ld_frag16(kp1 + 32), s1);

    // ---- online softmax (reductions within 16-lane groups of the C layout)
    const int key0 = kc + ln15;
    const int key1 = key0 + 16;
#pragma unroll
    for (int r = 0; r < 8; ++r) {
      const int m = q0 + mrow + r;
      float a = s0[r] * 0.125f;                     // 1/sqrt(64)
      float b = s1[r] * 0.125f;
      a = ((key0 <= m) && (key0 > m - WIN)) ? a : -1e30f;
      b = ((key1 <= m) && (key1 > m - WIN)) ? b : -1e30f;
      float cm = fmaxf(a, b);
      cm = fmaxf(cm, __shfl_xor(cm, 1));
      cm = fmaxf(cm, __shfl_xor(cm, 2));
      cm = fmaxf(cm, __shfl_xor(cm, 4));
      cm = fmaxf(cm, __shfl_xor(cm, 8));
      const float nm = fmaxf(rmax[r], cm);
      const float f  = __expf(rmax[r] - nm);        // 0 for -1e30 start: wipes junk
      rmax[r] = nm;
      const float e0 = __expf(a - nm);
      const float e1 = __expf(b - nm);
      float rs = e0 + e1;
      rs += __shfl_xor(rs, 1);  rs += __shfl_xor(rs, 2);
      rs += __shfl_xor(rs, 4);  rs += __shfl_xor(rs, 8);
      rsum[r] = rsum[r] * f + rs;
      o0[r] *= f; o1[r] *= f; o2[r] *= f; o3[r] *= f;
      Pt[wavein][mrow + r][ln15]      = f2bf(e0);
      Pt[wavein][mrow + r][ln15 + 16] = f2bf(e1);
    }

    // wave-internal LDS round trip: C-layout P -> A-layout fragment
    asm volatile("s_wait_dscnt 0" ::: "memory");
    v16bf pf = ld_fragA(&Pt[wavein][ln15][0], kbase);

    // ---- O += P(16x32) @ V(32x64) from the staged LDS V tile
    const bf16_t* vp = &Vl[buf][ln15][koff];
    o0 = wmma_bf16(pf, ld_frag16(vp),            o0);
    o1 = wmma_bf16(pf, ld_frag16(vp + 16 * VLP), o1);
    o2 = wmma_bf16(pf, ld_frag16(vp + 32 * VLP), o2);
    o3 = wmma_bf16(pf, ld_frag16(vp + 48 * VLP), o3);
  }

#pragma unroll
  for (int r = 0; r < 8; ++r) {
    const float inv = 1.0f / rsum[r];
    bf16_t* op = Oa + (size_t)(q0 + mrow + r) * (NH * DV) + (size_t)h * DV + ln15;
    op[0]  = f2bf(o0[r] * inv);
    op[16] = f2bf(o1[r] * inv);
    op[32] = f2bf(o2[r] * inv);
    op[48] = f2bf(o3[r] * inv);
  }
}

// ----------------------------------------------------------------- launcher
extern "C" void kernel_launch(void* const* d_in, const int* in_sizes, int n_in,
                              void* d_out, int out_size, void* d_ws, size_t ws_size,
                              hipStream_t stream) {
  (void)in_sizes; (void)n_in; (void)out_size; (void)ws_size;
  const float* hs  = (const float*)d_in[0];
  const int*   pos = (const int*)d_in[1];
  const float* Wq  = (const float*)d_in[2];
  const float* Wk  = (const float*)d_in[3];
  const float* Wv  = (const float*)d_in[4];
  const float* Wo  = (const float*)d_in[5];
  const float* qnw = (const float*)d_in[6];
  const float* knw = (const float*)d_in[7];
  float* out = (float*)d_out;

  char* ws = (char*)d_ws;
  const size_t MB = 1ull << 20;
  bf16_t* hs_b = (bf16_t*)(ws + 0);        //  8 MiB  [S,D]
  bf16_t* Wq_b = (bf16_t*)(ws + 8  * MB);  //  8 MiB
  bf16_t* Wk_b = (bf16_t*)(ws + 16 * MB);  //  2 MiB
  bf16_t* Wv_b = (bf16_t*)(ws + 18 * MB);  //  2 MiB
  bf16_t* Wo_b = (bf16_t*)(ws + 20 * MB);  //  8 MiB
  float*  Qt   = (float* )(ws + 28 * MB);  // 16 MiB  [S, NH*64]  f32
  float*  Kt   = (float* )(ws + 44 * MB);  //  4 MiB  [S, NKV*64] f32
  float*  Vtm  = (float* )(ws + 48 * MB);  //  4 MiB  [S, NKV*64] f32
  bf16_t* Qa   = (bf16_t*)(ws + 52 * MB);  //  8 MiB  [NH,S,64]
  bf16_t* Ka   = (bf16_t*)(ws + 60 * MB);  //  2 MiB  [NKV,S,64]
  bf16_t* Vt   = (bf16_t*)(ws + 62 * MB);  //  2 MiB  [NKV,64,S]
  bf16_t* Oa   = (bf16_t*)(ws + 64 * MB);  //  8 MiB  [S, NH*64]

  // 1) bf16 conversions
  cvt_f32_bf16<<<2048, 256, 0, stream>>>(hs, hs_b, S_LEN * D_DIM);
  cvt_f32_bf16<<<2048, 256, 0, stream>>>(Wq, Wq_b, NH  * DK * D_DIM);
  cvt_f32_bf16<<< 512, 256, 0, stream>>>(Wk, Wk_b, NKV * DK * D_DIM);
  cvt_f32_bf16<<< 512, 256, 0, stream>>>(Wv, Wv_b, NKV * DV * D_DIM);
  cvt_f32_bf16<<<2048, 256, 0, stream>>>(Wo, Wo_b, D_DIM * NH * DV);

  // 2) projections: waves = (M/32)*(N/64), 8 waves/block
  gemm_bf16_nt<<<256, 256, 0, stream>>>(hs_b, Wq_b, Qt,  S_LEN, NH  * DK, D_DIM);
  gemm_bf16_nt<<< 64, 256, 0, stream>>>(hs_b, Wk_b, Kt,  S_LEN, NKV * DK, D_DIM);
  gemm_bf16_nt<<< 64, 256, 0, stream>>>(hs_b, Wv_b, Vtm, S_LEN, NKV * DV, D_DIM);

  // 3) per-head RMSNorm + RoPE; V transpose+convert
  rmsnorm_rope<<<S_LEN * NH  / 8, 256, 0, stream>>>(Qt, qnw, pos, Qa, NH);
  rmsnorm_rope<<<S_LEN * NKV / 8, 256, 0, stream>>>(Kt, knw, pos, Ka, NKV);
  vconvert<<<NKV * DV * S_LEN / 256, 256, 0, stream>>>(Vtm, Vt);

  // 4) sliding-window flash attention: 1 head x 128 queries per block
  attn_swa<<<NH * (S_LEN / QB), 256, 0, stream>>>(Qa, Ka, Vt, Oa);

  // 5) output projection -> fp32 result
  gemm_bf16_nt<<<256, 256, 0, stream>>>(Oa, Wo_b, out, S_LEN, D_DIM, D_DIM);
}
